// TPUCapibaraByte_60902636257666
// MI455X (gfx1250) — matrix operand order repaired and checked by reading, not compiled, not run on hardware
//
#include <hip/hip_runtime.h>
#include <hip/hip_bf16.h>

typedef __attribute__((ext_vector_type(16))) __bf16 v16bf;
typedef __attribute__((ext_vector_type(8)))  __bf16 v8bf;
typedef __attribute__((ext_vector_type(8)))  float  v8f;

static constexpr int BB = 16;    // batch
static constexpr int SS = 1024;  // seq len
static constexpr int DD = 256;   // input dim
static constexpr int HH = 2048;  // hidden

// Workspace layout (all 256-byte aligned)
static constexpr size_t OFF_CNT = 0;                         // 4B sync counter
static constexpr size_t OFF_ST0 = 256;                       // bf16 state buf0 [16,2048]
static constexpr size_t OFF_ST1 = OFF_ST0 + 65536;           // bf16 state buf1
static constexpr size_t OFF_WS  = OFF_ST1 + 65536;           // packed W_state bf16 (8 MB)
static constexpr size_t OFF_WC  = OFF_WS + (size_t)HH*HH*2;  // packed W_conv bf16 (1 MB)
static constexpr size_t OFF_XN  = OFF_WC + (size_t)DD*HH*2;  // bf16 x_n [16384,256] (8 MB)

__device__ __forceinline__ float gelu_f(float x) {
    // jax.nn.gelu default (tanh approximation)
    float x3 = x * x * x;
    return 0.5f * x * (1.0f + tanhf(0.7978845608028654f * (x + 0.044715f * x3)));
}

// Pack a row-major fp32 [K,N] B-matrix into per-WMMA-fragment bf16 layout:
// P[((nt*KT + kt)*32 + lane)*16 + j] = W[kt*32 + (j/8)*16 + (lane/16)*8 + j%8, nt*16 + lane%16]
// so each lane's 16 B-fragment elements are one contiguous 32-byte load.
__global__ void pack_b_kernel(const float* __restrict__ W, __bf16* __restrict__ P,
                              int K, int N) {
    const int KT = K / 32;
    const size_t total = (size_t)K * N;
    for (size_t idx = (size_t)blockIdx.x * blockDim.x + threadIdx.x; idx < total;
         idx += (size_t)gridDim.x * blockDim.x) {
        int j    = (int)(idx & 15);
        int lane = (int)((idx >> 4) & 31);
        int kt   = (int)((idx >> 9) % KT);
        int nt   = (int)(idx / ((size_t)KT * 512));
        int row  = kt * 32 + (j >> 3) * 16 + (lane >> 4) * 8 + (j & 7);
        int col  = nt * 16 + (lane & 15);
        P[idx] = (__bf16)W[(size_t)row * N + col];
    }
}

__global__ void init_kernel(const float* __restrict__ init_state,
                            __bf16* __restrict__ st0, __bf16* __restrict__ st1,
                            unsigned* __restrict__ cnt) {
    int i = blockIdx.x * blockDim.x + threadIdx.x;
    if (i == 0) *cnt = 0u;
    if (i < BB * HH) {
        __bf16 v = (__bf16)init_state[i];
        st0[i] = v;
        st1[i] = v;
    }
}

// One 256-thread block per (b,s) row: LayerNorm over D=256, emit bf16.
__global__ void ln_kernel(const float* __restrict__ x, const float* __restrict__ scale,
                          const float* __restrict__ bias, __bf16* __restrict__ xn) {
    __shared__ float red[256];
    const int row = blockIdx.x;
    const int tid = threadIdx.x;
    float v = x[(size_t)row * DD + tid];
    red[tid] = v;
    __syncthreads();
    for (int s = 128; s > 0; s >>= 1) {
        if (tid < s) red[tid] += red[tid + s];
        __syncthreads();
    }
    float mean = red[0] * (1.0f / DD);
    __syncthreads();
    float d = v - mean;
    red[tid] = d * d;
    __syncthreads();
    for (int s = 128; s > 0; s >>= 1) {
        if (tid < s) red[tid] += red[tid + s];
        __syncthreads();
    }
    float var = red[0] * (1.0f / DD);
    float y = d * rsqrtf(var + 1e-6f) * scale[tid] + bias[tid];
    xn[(size_t)row * DD + tid] = (__bf16)y;
}

// xw = x_n @ W_conv  (M=16384, K=256, N=2048), fp32 result into all_states region.
// One wave per 16x16 output tile; 8 waves per 256-thread block.
__global__ void conv_gemm_kernel(const __bf16* __restrict__ xn,
                                 const __bf16* __restrict__ Wp,
                                 float* __restrict__ out_states) {
    const int wave = blockIdx.x * (blockDim.x >> 5) + (threadIdx.x >> 5);
    const int lane = threadIdx.x & 31;
    const int mt = wave >> 7;     // 0..1023
    const int nt = wave & 127;    // 0..127
    const int row  = lane & 15;
    const int half = lane >> 4;
    v8f c = {};
    const __bf16* arow  = xn + (size_t)(mt * 16 + row) * DD + half * 8;
    const __bf16* bbase = Wp + (size_t)nt * (DD / 32) * 512 + (size_t)lane * 16;
#pragma unroll
    for (int kt = 0; kt < DD / 32; ++kt) {
        v8bf alo = *(const v8bf*)(arow + kt * 32);
        v8bf ahi = *(const v8bf*)(arow + kt * 32 + 16);
        v16bf a;
#pragma unroll
        for (int i = 0; i < 8; ++i) { a[i] = alo[i]; a[i + 8] = ahi[i]; }
        v16bf b = *(const v16bf*)(bbase + (size_t)kt * 512);
        c = __builtin_amdgcn_wmma_f32_16x16x32_bf16(false, a, false, b, (short)0, c,
                                                    false, false);
    }
#pragma unroll
    for (int r = 0; r < 8; ++r) {
        int m = mt * 16 + r + half * 8;
        int n = nt * 16 + (lane & 15);
        out_states[(size_t)m * HH + n] = c[r];
    }
}

// Persistent scan kernel: 32 blocks x 4 waves = 128 waves; wave w owns hidden
// columns [16w, 16w+16) and keeps its entire W_state column slice resident in
// VGPRs (64 fragments = 512 VGPRs/lane). Each step the block stages the bf16
// state into LDS once; A-fragments then come from LDS. Device-wide barrier
// between the 1024 dependent steps.
__global__ void __launch_bounds__(128, 1) scan_kernel(
        const float* __restrict__ init_state, const __bf16* __restrict__ Wp,
        __bf16* __restrict__ st0, __bf16* __restrict__ st1,
        float* __restrict__ states, float* __restrict__ final_out,
        unsigned* __restrict__ cnt) {
    __shared__ __bf16 slds[BB * HH];   // 64 KB staged state
    const unsigned NWG = 32;
    const int wave = blockIdx.x * 4 + (threadIdx.x >> 5);
    const int lane = threadIdx.x & 31;
    const int nt   = wave;              // 0..127 hidden-column tile
    const int row  = lane & 15;
    const int half = lane >> 4;
    const int col  = nt * 16 + row;     // hidden index owned by this lane

    // ---- hoist this wave's W_state slice into registers (reused 1024x) ----
    const __bf16* bbase = Wp + (size_t)nt * 64 * 512 + (size_t)lane * 16;
    v16bf wreg[HH / 32];
#pragma unroll
    for (int kt = 0; kt < HH / 32; ++kt)
        wreg[kt] = *(const v16bf*)(bbase + (size_t)kt * 512);

    for (int t = 0; t < SS; ++t) {
        const __bf16* cur = (t & 1) ? st1 : st0;
        __bf16* nxt = (t & 1) ? st0 : st1;

        // ---- cooperative stage: state [16,2048] bf16 -> LDS (512 B/thread) ----
        {
            const v8bf* __restrict__ src = (const v8bf*)cur;
            v8bf* dst = (v8bf*)slds;
#pragma unroll
            for (int i = 0; i < (BB * HH / 8) / 128; ++i) {
                int idx = i * 128 + threadIdx.x;
                dst[idx] = src[idx];
            }
        }
        __syncthreads();

        const __bf16* arow = slds + (size_t)row * HH + half * 8;
        v8f c = {};
#pragma unroll
        for (int kt = 0; kt < HH / 32; ++kt) {
            v8bf alo = *(const v8bf*)(arow + kt * 32);
            v8bf ahi = *(const v8bf*)(arow + kt * 32 + 16);
            v16bf a;
#pragma unroll
            for (int i = 0; i < 8; ++i) { a[i] = alo[i]; a[i + 8] = ahi[i]; }
            c = __builtin_amdgcn_wmma_f32_16x16x32_bf16(false, a, false, wreg[kt],
                                                        (short)0, c, false, false);
        }

#pragma unroll
        for (int r = 0; r < 8; ++r) {
            int m = r + half * 8;                         // batch index
            size_t o = ((size_t)m * SS + t) * HH + col;   // all_states[b,t,h]
            float xw = states[o];                         // precomputed projection
            float sold = (t == 0) ? init_state[(size_t)m * HH + col]
                                  : states[o - HH];       // all_states[b,t-1,h]
            float g = gelu_f(c[r] + xw);
            float snew = 0.9f * sold + 0.1f * g;
            states[o] = snew;
            nxt[(size_t)m * HH + col] = (__bf16)snew;
            if (t == SS - 1) final_out[(size_t)m * HH + col] = snew;
        }

        // ---- device-wide barrier (monotone arrival counter) ----
        __threadfence();
        __syncthreads();
        if (threadIdx.x == 0) {
            __hip_atomic_fetch_add(cnt, 1u, __ATOMIC_ACQ_REL,
                                   __HIP_MEMORY_SCOPE_AGENT);
            unsigned target = NWG * (unsigned)(t + 1);
            while (__hip_atomic_load(cnt, __ATOMIC_ACQUIRE,
                                     __HIP_MEMORY_SCOPE_AGENT) < target) {
                __builtin_amdgcn_s_sleep(1);
            }
        }
        __syncthreads();
        __threadfence();
    }
}

extern "C" void kernel_launch(void* const* d_in, const int* in_sizes, int n_in,
                              void* d_out, int out_size, void* d_ws, size_t ws_size,
                              hipStream_t stream) {
    const float* x   = (const float*)d_in[0];
    const float* s0  = (const float*)d_in[1];
    const float* Ws  = (const float*)d_in[2];
    const float* Wc  = (const float*)d_in[3];
    const float* lns = (const float*)d_in[4];
    const float* lnb = (const float*)d_in[5];

    float* out       = (float*)d_out;
    float* final_out = out;                      // [B,H]
    float* states    = out + (size_t)BB * HH;    // [B,S,H] (xw scratch, then states)

    char* ws = (char*)d_ws;
    unsigned* cnt = (unsigned*)(ws + OFF_CNT);
    __bf16* st0   = (__bf16*)(ws + OFF_ST0);
    __bf16* st1   = (__bf16*)(ws + OFF_ST1);
    __bf16* Wsp   = (__bf16*)(ws + OFF_WS);
    __bf16* Wcp   = (__bf16*)(ws + OFF_WC);
    __bf16* xn    = (__bf16*)(ws + OFF_XN);

    // 1) pack weights to WMMA B-fragment layout (bf16)
    pack_b_kernel<<<2048, 256, 0, stream>>>(Ws, Wsp, HH, HH);
    pack_b_kernel<<<512, 256, 0, stream>>>(Wc, Wcp, DD, HH);
    // 2) zero sync counter + seed bf16 state ping-pong buffers
    init_kernel<<<(BB * HH + 255) / 256, 256, 0, stream>>>(s0, st0, st1, cnt);
    // 3) layernorm -> bf16
    ln_kernel<<<BB * SS, 256, 0, stream>>>(x, lns, lnb, xn);
    // 4) xw = x_n @ W_conv into all_states region (one wave per 16x16 tile)
    conv_gemm_kernel<<<(BB * SS / 16) * (HH / 16) / 8, 256, 0, stream>>>(xn, Wcp,
                                                                         states);
    // 5) persistent recurrent scan: weights in VGPRs, state staged via LDS
    scan_kernel<<<32, 128, 0, stream>>>(s0, Wsp, st0, st1, states, final_out, cnt);
}
